// AdaptiveAttention_16260746183148
// MI455X (gfx1250) — matrix operand order repaired and checked by reading, not compiled, not run on hardware
//
#include <hip/hip_runtime.h>
#include <hip/hip_bf16.h>
#include <math.h>

#define HDIM 1024
#define NBATCH 256
#define SEQ 196
#define SEQ1 197
#define NROWS (NBATCH * SEQ)          // 50176 flattened (b, j) rows
#define LDSTRIDE 136                  // 128 bf16 + 8 bf16 pad (272 B, 16B-aligned)

typedef __bf16 bf16_t;
typedef bf16_t v16bf __attribute__((ext_vector_type(16)));
typedef bf16_t v8bf  __attribute__((ext_vector_type(8)));
typedef float  v8f   __attribute__((ext_vector_type(8)));
typedef unsigned short v8us __attribute__((ext_vector_type(8)));
typedef unsigned int   v4u  __attribute__((ext_vector_type(4)));
typedef int            v4i  __attribute__((ext_vector_type(4)));
typedef int            v8i  __attribute__((ext_vector_type(8)));

#if defined(__has_builtin)
#if __has_builtin(__builtin_amdgcn_tensor_load_to_lds) && \
    __has_builtin(__builtin_amdgcn_s_wait_tensorcnt)
#define HAVE_TDM 1
#endif
#endif

// fp32 -> bf16 bits (round to nearest even)
__device__ __forceinline__ unsigned short f2bf(float x) {
    union { float f; unsigned u; } v; v.f = x;
    unsigned r = v.u + 0x7FFFu + ((v.u >> 16) & 1u);
    return (unsigned short)(r >> 16);
}

// ---------------------------------------------------------------------------
// Streaming fp32 -> bf16 conversion of v (done once; feeds the TDM path).
// ---------------------------------------------------------------------------
__global__ __launch_bounds__(256)
void conv_v_kernel(const float* __restrict__ src, unsigned short* __restrict__ dst) {
    size_t i = ((size_t)blockIdx.x * 256 + threadIdx.x) * 8;
    float4 a = *(const float4*)&src[i];
    float4 b = *(const float4*)&src[i + 4];
    v8us o;
    o[0] = f2bf(a.x); o[1] = f2bf(a.y); o[2] = f2bf(a.z); o[3] = f2bf(a.w);
    o[4] = f2bf(b.x); o[5] = f2bf(b.y); o[6] = f2bf(b.z); o[7] = f2bf(b.w);
    *(v8us*)&dst[i] = o;
}

// ---------------------------------------------------------------------------
// Pack fp32 W (K x N, row-major) into WMMA B-fragment order (bf16).
// Lane l, value i -> element (k = kt*32 + (l/16)*16 + i, n = nt*16 + l%16),
// stored so each lane reads 16 contiguous bf16 (32 B) per fragment.
// ---------------------------------------------------------------------------
__global__ void pack_w_kernel(const float* __restrict__ W,
                              unsigned short* __restrict__ dst,
                              int K, int N) {
    int tid = blockIdx.x * blockDim.x + threadIdx.x;
    int total = (N / 16) * (K / 32) * 32;
    if (tid >= total) return;
    int l  = tid & 31;
    int kt = (tid >> 5) % (K / 32);
    int nt = tid / (32 * (K / 32));
    int n  = nt * 16 + (l & 15);
    int kbase = kt * 32 + ((l >> 4) << 4);
    unsigned short* out = dst + (size_t)tid * 16;
#pragma unroll
    for (int i = 0; i < 16; ++i)
        out[i] = f2bf(W[(size_t)(kbase + i) * N + n]);
}

// A-fragment load helper (16-bit A 16x32, contiguous k = 16*(l/16) + i) from padded LDS tile.
__device__ __forceinline__ v16bf load_afrag(const unsigned short* sA,
                                            int mt, int l, int ks) {
    int abase = (mt * 16 + (l & 15)) * LDSTRIDE + ks * 32 + ((l >> 4) << 4);
    v8bf alo = *(const v8bf*)&sA[abase];
    v8bf ahi = *(const v8bf*)&sA[abase + 8];
    return __builtin_shufflevector(alo, ahi,
        0, 1, 2, 3, 4, 5, 6, 7, 8, 9, 10, 11, 12, 13, 14, 15);
}

#if HAVE_TDM
// Issue one 64x128 bf16 tile DMA: global (row stride 1024) -> LDS (padded
// rows: pad_interval = 64 dwords = one 256 B row, pad_amount = 4 dwords).
__device__ __forceinline__ void tdm_issue(const unsigned short* gsrc,
                                          unsigned short* lds_dst) {
    unsigned long long ga = (unsigned long long)(const void*)gsrc;
    unsigned lds_off = (unsigned)(unsigned long long)(void*)lds_dst;
    v4u g0;
    g0[0] = 1u;                                     // count = 1 descriptor
    g0[1] = lds_off;                                // lds_addr
    g0[2] = (unsigned)(ga & 0xFFFFFFFFu);
    g0[3] = (unsigned)((ga >> 32) & 0x01FFFFFFu) | (2u << 30); // type = 2
    v8i g1;
    g1[0] = (int)((1u << 16) | (1u << 20) | (5u << 22) | (3u << 25));
    g1[1] = (int)(128u << 16);                      // tensor_dim0 = 128
    g1[2] = (int)(64u << 16);                       // tensor_dim1 = 64
    g1[3] = (int)(128u << 16);                      // tile_dim0 = 128
    g1[4] = (int)64u;                               // tile_dim1 = 64
    g1[5] = (int)1024u;                             // tensor_dim0_stride
    g1[6] = 0;
    g1[7] = 0;
    v4i zero4 = {0, 0, 0, 0};
#if __clang_major__ >= 23
    v8i zero8 = {0, 0, 0, 0, 0, 0, 0, 0};
    __builtin_amdgcn_tensor_load_to_lds(g0, g1, zero4, zero4, zero8, 0);
#else
    __builtin_amdgcn_tensor_load_to_lds(g0, g1, zero4, zero4, 0);
#endif
}
#endif

// ---------------------------------------------------------------------------
// Generic C = act(A @ W + bias), K = 1024, bf16 WMMA, f32 accumulate.
// 256 threads (8 waves) -> 64 rows x 256 cols tile. A is fp32 (small inputs).
// ---------------------------------------------------------------------------
__global__ __launch_bounds__(256)
void wmma_gemm_act(const float* __restrict__ A, int lda,
                   const unsigned short* __restrict__ Wp,
                   const float* __restrict__ bias,
                   float* __restrict__ C, int ldc, int do_tanh) {
    __shared__ unsigned short sA[64 * LDSTRIDE];
    const int t = threadIdx.x;
    const int w = t >> 5;
    const int l = t & 31;
    const int rowbase = blockIdx.x * 64;
    const int colbase = blockIdx.y * 256;
    const int mt = w & 3;
    const int nhalf = w >> 2;

    v8f acc[8];
#pragma unroll
    for (int s = 0; s < 8; ++s)
#pragma unroll
        for (int g = 0; g < 8; ++g) acc[s][g] = 0.0f;

    for (int kc = 0; kc < 8; ++kc) {
        __syncthreads();
#pragma unroll
        for (int i = 0; i < 32; ++i) {           // 8192 elems / 256 threads
            int e = t + i * 256;
            int r = e >> 7, k = e & 127;
            sA[r * LDSTRIDE + k] =
                f2bf(A[(size_t)(rowbase + r) * lda + kc * 128 + k]);
        }
        __syncthreads();
#pragma unroll
        for (int ks = 0; ks < 4; ++ks) {
            v16bf afrag = load_afrag(sA, mt, l, ks);
            int ktg = kc * 4 + ks;
#pragma unroll
            for (int s = 0; s < 8; ++s) {
                int ntg = (colbase >> 4) + nhalf * 8 + s;
                v16bf bfrag = *(const v16bf*)(Wp +
                    ((size_t)(ntg * 32 + ktg) * 32 + l) * 16);
                acc[s] = __builtin_amdgcn_wmma_f32_16x16x32_bf16(
                    false, afrag, false, bfrag, (short)0, acc[s], false, false);
            }
        }
    }
#pragma unroll
    for (int s = 0; s < 8; ++s) {
        int col = colbase + (nhalf * 8 + s) * 16 + (l & 15);
        float bv = bias[col];
#pragma unroll
        for (int g = 0; g < 8; ++g) {
            int row = rowbase + mt * 16 + g + ((l >> 4) << 3);
            float val = acc[s][g] + bv;
            if (do_tanh) val = tanhf(val);
            C[(size_t)row * ldc + col] = val;
        }
    }
}

// ---------------------------------------------------------------------------
// Shared epilogue: fold bias + _h + tanh + W_z dot into LDS z-accumulators.
// ---------------------------------------------------------------------------
__device__ __forceinline__ void z_epilogue(const v8f* acc, float* zbuf,
                                           int nc, int nhalf, int mt, int l,
                                           int rowbase,
                                           const float* __restrict__ b_v,
                                           const float* __restrict__ Hh,
                                           const float* __restrict__ Wz) {
#pragma unroll
    for (int s = 0; s < 8; ++s) {
        int col = nc * 256 + (nhalf * 8 + s) * 16 + (l & 15);
        float bv = b_v[col];
        float wz = Wz[col];
#pragma unroll
        for (int g = 0; g < 8; ++g) {
            int row_l = mt * 16 + g + ((l >> 4) << 3);
            int row_g = rowbase + row_l;
            int b = row_g / SEQ;
            float x = acc[s][g] + bv + Hh[(size_t)b * HDIM + col];
            atomicAdd(&zbuf[row_l], tanhf(x) * wz);
        }
    }
}

// ---------------------------------------------------------------------------
// Big fused kernel, TDM variant with DOUBLE-BUFFERED tensor DMA:
// wave 0 issues the next chunk's tensor_load_to_lds into the ping-pong
// buffer, then waits tensorcnt<=1 (TDM ops are in-order per wave, so the
// current chunk is guaranteed resident) -> the DMA of chunk kc+1 overlaps
// the 32 WMMAs of chunk kc across all 8 waves.
//   z[b, j] = tanh(v_row @ W_v + b_v + _h[b]) . W_z + b_z
// ---------------------------------------------------------------------------
__global__ __launch_bounds__(256)
void wmma_fused_zv_tdm(const unsigned short* __restrict__ vbf,
                       const unsigned short* __restrict__ Wvp,
                       const float* __restrict__ b_v,
                       const float* __restrict__ Hh,
                       const float* __restrict__ Wz,
                       const float* __restrict__ b_z,
                       float* __restrict__ z) {
    __shared__ __align__(256) unsigned short sA[2][64 * LDSTRIDE];
    __shared__ float zbuf[64];
    const int t = threadIdx.x;
    const int w = t >> 5;
    const int l = t & 31;
    const int rowbase = blockIdx.x * 64;
    const int mt = w & 3;
    const int nhalf = w >> 2;
    const unsigned short* vrow = vbf + (size_t)rowbase * HDIM;

    if (t < 64) zbuf[t] = 0.0f;

    for (int nc = 0; nc < 4; ++nc) {
        v8f acc[8];
#pragma unroll
        for (int s = 0; s < 8; ++s)
#pragma unroll
            for (int g = 0; g < 8; ++g) acc[s][g] = 0.0f;

#if HAVE_TDM
        // preload chunk 0 (buffer 0 was last read two chunks ago -> free)
        if (w == 0) tdm_issue(vrow, sA[0]);
#endif
        for (int kc = 0; kc < 8; ++kc) {
            __syncthreads();    // everyone done with the buffer being refilled
#if HAVE_TDM
            if (w == 0) {
                if (kc < 7) {
                    tdm_issue(vrow + (kc + 1) * 128, sA[(kc + 1) & 1]);
                    __builtin_amdgcn_s_wait_tensorcnt(1); // chunk kc resident
                } else {
                    __builtin_amdgcn_s_wait_tensorcnt(0);
                }
            }
#else
#pragma unroll
            for (int i = 0; i < 32; ++i) {
                int e = t + i * 256;
                int r = e >> 7, k = e & 127;
                sA[kc & 1][r * LDSTRIDE + k] =
                    vbf[(size_t)(rowbase + r) * HDIM + kc * 128 + k];
            }
#endif
            __syncthreads();    // tile kc visible to all waves
            const unsigned short* tile = sA[kc & 1];
#pragma unroll
            for (int ks = 0; ks < 4; ++ks) {
                v16bf afrag = load_afrag(tile, mt, l, ks);
                int ktg = kc * 4 + ks;
#pragma unroll
                for (int s = 0; s < 8; ++s) {
                    int ntg = (nc * 16) + nhalf * 8 + s;
                    v16bf bfrag = *(const v16bf*)(Wvp +
                        ((size_t)(ntg * 32 + ktg) * 32 + l) * 16);
                    acc[s] = __builtin_amdgcn_wmma_f32_16x16x32_bf16(
                        false, afrag, false, bfrag, (short)0, acc[s],
                        false, false);
                }
            }
        }
        z_epilogue(acc, zbuf, nc, nhalf, mt, l, rowbase, b_v, Hh, Wz);
    }
    __syncthreads();
    if (t < 64) {
        int rg = rowbase + t;
        int b = rg / SEQ, j = rg % SEQ;
        z[b * SEQ1 + j] = zbuf[t] + b_z[0];
    }
}

// ---------------------------------------------------------------------------
// Big fused kernel, fp32-source fallback (used when workspace cannot hold the
// bf16 copy of v): stages + converts v in-kernel.
// ---------------------------------------------------------------------------
__global__ __launch_bounds__(256)
void wmma_fused_zv(const float* __restrict__ v,
                   const unsigned short* __restrict__ Wvp,
                   const float* __restrict__ b_v,
                   const float* __restrict__ Hh,
                   const float* __restrict__ Wz,
                   const float* __restrict__ b_z,
                   float* __restrict__ z) {
    __shared__ unsigned short sA[64 * LDSTRIDE];
    __shared__ float zbuf[64];
    const int t = threadIdx.x;
    const int w = t >> 5;
    const int l = t & 31;
    const int rowbase = blockIdx.x * 64;
    const int mt = w & 3;
    const int nhalf = w >> 2;

    if (t < 64) zbuf[t] = 0.0f;

    for (int nc = 0; nc < 4; ++nc) {
        v8f acc[8];
#pragma unroll
        for (int s = 0; s < 8; ++s)
#pragma unroll
            for (int g = 0; g < 8; ++g) acc[s][g] = 0.0f;

        for (int kc = 0; kc < 8; ++kc) {
            __syncthreads();
#pragma unroll
            for (int i = 0; i < 32; ++i) {
                int e = t + i * 256;
                int r = e >> 7, k = e & 127;
                sA[r * LDSTRIDE + k] =
                    f2bf(v[(size_t)(rowbase + r) * HDIM + kc * 128 + k]);
            }
            __syncthreads();
#pragma unroll
            for (int ks = 0; ks < 4; ++ks) {
                v16bf afrag = load_afrag(sA, mt, l, ks);
                int ktg = kc * 4 + ks;
#pragma unroll
                for (int s = 0; s < 8; ++s) {
                    int ntg = (nc * 16) + nhalf * 8 + s;
                    v16bf bfrag = *(const v16bf*)(Wvp +
                        ((size_t)(ntg * 32 + ktg) * 32 + l) * 16);
                    acc[s] = __builtin_amdgcn_wmma_f32_16x16x32_bf16(
                        false, afrag, false, bfrag, (short)0, acc[s],
                        false, false);
                }
            }
        }
        z_epilogue(acc, zbuf, nc, nhalf, mt, l, rowbase, b_v, Hh, Wz);
    }
    __syncthreads();
    if (t < 64) {
        int rg = rowbase + t;
        int b = rg / SEQ, j = rg % SEQ;
        z[b * SEQ1 + j] = zbuf[t] + b_z[0];
    }
}

// z for the appended s-row: z[b, S] = tanh(_s + _h) . W_z + b_z
__global__ __launch_bounds__(256)
void zs_kernel(const float* __restrict__ Ss, const float* __restrict__ Hh,
               const float* __restrict__ Wz, const float* __restrict__ b_z,
               float* __restrict__ z) {
    __shared__ float red[256];
    int b = blockIdx.x, t = threadIdx.x;
    float p = 0.0f;
    for (int n = t; n < HDIM; n += 256)
        p += tanhf(Ss[(size_t)b * HDIM + n] + Hh[(size_t)b * HDIM + n]) * Wz[n];
    red[t] = p;
    __syncthreads();
    for (int st = 128; st > 0; st >>= 1) {
        if (t < st) red[t] += red[t + st];
        __syncthreads();
    }
    if (t == 0) z[b * SEQ1 + SEQ] = red[0] + b_z[0];
}

// Per-batch softmax over 197 logits + context c = sum_j a_j * vc_j
__global__ __launch_bounds__(256)
void softmax_ctx_kernel(const float* __restrict__ z, const float* __restrict__ v,
                        const float* __restrict__ hs,
                        float* __restrict__ out_a, float* __restrict__ out_c) {
    __shared__ float zl[SEQ1];
    __shared__ float red[256];
    int b = blockIdx.x, t = threadIdx.x;
    if (t < SEQ1) zl[t] = z[b * SEQ1 + t];
    __syncthreads();
    red[t] = (t < SEQ1) ? zl[t] : -1e30f;
    __syncthreads();
    for (int st = 128; st > 0; st >>= 1) {
        if (t < st) red[t] = fmaxf(red[t], red[t + st]);
        __syncthreads();
    }
    float mx = red[0];
    __syncthreads();
    float e = (t < SEQ1) ? __expf(zl[t] - mx) : 0.0f;
    red[t] = e;
    __syncthreads();
    for (int st = 128; st > 0; st >>= 1) {
        if (t < st) red[t] += red[t + st];
        __syncthreads();
    }
    float inv = 1.0f / red[0];
    __syncthreads();
    if (t < SEQ1) {
        float a = e * inv;
        zl[t] = a;
        out_a[b * SEQ1 + t] = a;
    }
    __syncthreads();
    int col = t * 4;
    float aS = zl[SEQ];
    const float* sb = hs + (size_t)b * 2048 + HDIM;   // s = hs[:, H:]
    float4 acc;
    acc.x = aS * sb[col + 0];
    acc.y = aS * sb[col + 1];
    acc.z = aS * sb[col + 2];
    acc.w = aS * sb[col + 3];
    const float* vb = v + (size_t)b * SEQ * HDIM;
    for (int j = 0; j < SEQ; ++j) {
        float aj = zl[j];
        float4 vv = *(const float4*)&vb[(size_t)j * HDIM + col];
        acc.x += aj * vv.x;
        acc.y += aj * vv.y;
        acc.z += aj * vv.z;
        acc.w += aj * vv.w;
    }
    *(float4*)&out_c[(size_t)b * HDIM + col] = acc;
}

extern "C" void kernel_launch(void* const* d_in, const int* in_sizes, int n_in,
                              void* d_out, int out_size, void* d_ws, size_t ws_size,
                              hipStream_t stream) {
    (void)in_sizes; (void)n_in; (void)out_size;
    const float* v    = (const float*)d_in[0];
    const float* h    = (const float*)d_in[1];
    const float* W_fc = (const float*)d_in[2];
    const float* b_fc = (const float*)d_in[3];
    const float* W_v  = (const float*)d_in[4];
    const float* b_v  = (const float*)d_in[5];
    const float* W_h  = (const float*)d_in[6];
    const float* b_h  = (const float*)d_in[7];
    const float* W_s  = (const float*)d_in[8];
    const float* b_s  = (const float*)d_in[9];
    const float* W_z  = (const float*)d_in[10];
    const float* b_z  = (const float*)d_in[11];

    char* p = (char*)d_ws;
    unsigned short* Wfc_p = (unsigned short*)p; p += (size_t)1024 * 2048 * 2;
    unsigned short* Wv_p  = (unsigned short*)p; p += (size_t)1024 * 1024 * 2;
    unsigned short* Wh_p  = (unsigned short*)p; p += (size_t)1024 * 1024 * 2;
    unsigned short* Ws_p  = (unsigned short*)p; p += (size_t)1024 * 1024 * 2;
    float* hs  = (float*)p; p += (size_t)NBATCH * 2048 * 4;
    float* Hh  = (float*)p; p += (size_t)NBATCH * 1024 * 4;
    float* Ss  = (float*)p; p += (size_t)NBATCH * 1024 * 4;
    float* zbu = (float*)p; p += (size_t)NBATCH * SEQ1 * 4;
    unsigned short* vbf = (unsigned short*)p;
    size_t base_need = (size_t)(p - (char*)d_ws);
    size_t vbf_bytes = (size_t)NROWS * HDIM * 2;
    bool use_tdm_path = (ws_size >= base_need + vbf_bytes);

    // 1) pack weights to bf16 WMMA fragment order
    pack_w_kernel<<<512, 256, 0, stream>>>(W_fc, Wfc_p, 1024, 2048);
    pack_w_kernel<<<256, 256, 0, stream>>>(W_v,  Wv_p,  1024, 1024);
    pack_w_kernel<<<256, 256, 0, stream>>>(W_h,  Wh_p,  1024, 1024);
    pack_w_kernel<<<256, 256, 0, stream>>>(W_s,  Ws_p,  1024, 1024);

    // 2) hs = tanh(h @ W_fc + b_fc)
    wmma_gemm_act<<<dim3(4, 8), 256, 0, stream>>>(h, 1024, Wfc_p, b_fc,
                                                  hs, 2048, 1);
    // 3) _h = h1 @ W_h + b_h ; _s = s @ W_s + b_s
    wmma_gemm_act<<<dim3(4, 4), 256, 0, stream>>>(hs, 2048, Wh_p, b_h,
                                                  Hh, 1024, 0);
    wmma_gemm_act<<<dim3(4, 4), 256, 0, stream>>>(hs + 1024, 2048, Ws_p, b_s,
                                                  Ss, 1024, 0);
    // 4) z[:, S] from the s-row
    zs_kernel<<<NBATCH, 256, 0, stream>>>(Ss, Hh, W_z, b_z, zbu);
    // 5) big fused GEMM -> z[:, 0..S)
    if (use_tdm_path) {
        conv_v_kernel<<<(NROWS * HDIM) / (256 * 8), 256, 0, stream>>>(v, vbf);
        wmma_fused_zv_tdm<<<NROWS / 64, 256, 0, stream>>>(vbf, Wv_p, b_v, Hh,
                                                          W_z, b_z, zbu);
    } else {
        wmma_fused_zv<<<NROWS / 64, 256, 0, stream>>>(v, Wv_p, b_v, Hh,
                                                      W_z, b_z, zbu);
    }
    // 6) softmax + context
    float* out_a = (float*)d_out;
    float* out_c = out_a + (size_t)NBATCH * SEQ1;
    softmax_ctx_kernel<<<NBATCH, 256, 0, stream>>>(zbu, v, hs, out_a, out_c);
}
